// GMGRU_20048907338172
// MI455X (gfx1250) — matrix operand order repaired and checked by reading, not compiled, not run on hardware
//
#include <hip/hip_runtime.h>

#define DD  33
#define TT  49
#define HIDN 64
#define QQ  64
#define H3  192
#define HSZ (TT*DD*HIDN)      // H_s element count; x_pred follows in d_out
#define NFRAG (6*DD*4*2)      // 6 tensors x 33 d x 4 mtiles x 2 ktiles = 1584
                              // each fragment: 32 lanes x 16 halfs = 1KB

typedef __attribute__((ext_vector_type(16))) _Float16 v16h;
typedef __attribute__((ext_vector_type(8)))  float    v8f;

struct Params {
  const float *data, *h0, *H0, *gh_w, *gh_b, *gst_w, *gst_b,
      *gru_wih, *gru_whh, *gru_bih, *gru_bhh,
      *wu0_w, *wu0_b, *wu1_w, *wu1_b, *wu_w, *wu_b,
      *wh_w, *wh_b, *xst_w, *xst_b, *wrg_w, *wrs,
      *tWz, *tUzx, *tUzh, *tWr, *tUrx, *tUrh, *tWh, *tUhx, *tUhh,
      *tbz, *tbr, *tbh;
  const void* wpk;   // pre-packed f16 A-fragments in d_ws (or nullptr)
  float* out;
};

// ISA 16-bit A layout (16x32 tile): lane l -> row (l&15);
// kbase = (l<16?0:8); element e -> K = kbase + (e<8 ? e : 8+e).
__device__ __forceinline__ v16h load_A_frag_f32(const float* __restrict__ W, int row0, int k0) {
  unsigned lane = threadIdx.x & 31u;
  const float* rowp = W + (row0 + (int)(lane & 15u)) * HIDN;
  int kbase = k0 + ((lane < 16u) ? 0 : 8);
  v16h a;
#pragma unroll
  for (int e = 0; e < 16; ++e) {
    int K = kbase + (e < 8 ? e : 8 + e);
    a[e] = (_Float16)rowp[K];
  }
  return a;
}

// B-fragment (32x16): vector v in column 0 only. kbase = k0 + (lane<16?0:8);
// element e -> K = kbase + (e<8 ? e : 8+e); only lanes with (lane&15)==0 are non-zero.
__device__ __forceinline__ v16h load_B_vec(const float* __restrict__ v, int k0) {
  unsigned lane = threadIdx.x & 31u;
  v16h b;
#pragma unroll
  for (int e = 0; e < 16; ++e) b[e] = (_Float16)0.0f;
  if ((lane & 15u) == 0u) {
    int kb = k0 + ((lane < 16u) ? 0 : 8);
#pragma unroll
    for (int e = 0; e < 16; ++e) b[e] = (_Float16)v[kb + (e < 8 ? e : 8 + e)];
  }
  return b;
}

__device__ __forceinline__ float sigm(float x) { return 1.0f / (1.0f + __expf(-x)); }

// -------- Kernel 1: pack the six (D,64,64) f32 tensors into per-lane f16
// WMMA A-fragments. Fragment f = ((tensor*DD + d)*4 + mt)*2 + kt; within a
// fragment, lane's 16 halfs are contiguous (32B) -> single vector load later.
__global__ __launch_bounds__(256) void gmgru_pack_wf16(
    const float* __restrict__ tWz, const float* __restrict__ tUzh,
    const float* __restrict__ tWr, const float* __restrict__ tUrh,
    const float* __restrict__ tWh, const float* __restrict__ tUhh,
    _Float16* __restrict__ out) {
  int gid = blockIdx.x * 256 + threadIdx.x;
  if (gid >= NFRAG * 32) return;
  int lane = gid & 31;
  int f    = gid >> 5;
  int kt   = f & 1;
  int mt   = (f >> 1) & 3;
  int rest = f >> 3;
  int d      = rest % DD;
  int tensor = rest / DD;
  const float* Ws[6] = { tWz, tUzh, tWr, tUrh, tWh, tUhh };
  const float* W = Ws[tensor] + d * HIDN * HIDN;
  int row   = mt * 16 + (lane & 15);
  int kbase = kt * 32 + ((lane < 16) ? 0 : 8);
  v16h a;
#pragma unroll
  for (int e = 0; e < 16; ++e) {
    int K = kbase + (e < 8 ? e : 8 + e);
    a[e] = (_Float16)W[row * HIDN + K];
  }
  *(v16h*)(out + (size_t)f * 512 + (size_t)lane * 16) = a;
}

// -------- Kernel 2: persistent single-workgroup sequential RNN --------
__global__ __launch_bounds__(1024) void gmgru_persistent(Params p) {
  const int tid = threadIdx.x;

  __shared__ float sH[DD*HIDN];
  __shared__ float sZ[DD*HIDN], sR[DD*HIDN], sWh[DD*HIDN], sUh[DD*HIDN];
  __shared__ float sh[HIDN], hist[3][HIDN], hcat[H3];
  __shared__ float s_x[DD], s_m[DD], s_dl[DD];
  __shared__ float s_xst[DD], s_xstc[DD];
  __shared__ float s_gamma[HIDN];
  __shared__ float s_gi[H3], s_gh[H3];
  __shared__ float s_u[QQ];
  __shared__ float s_xgt[DD], s_a[DD], s_b[DD];

  for (int e = tid; e < DD*HIDN; e += blockDim.x) sH[e] = p.H0[e];
  if (tid < HIDN) sh[tid] = p.h0[tid];
  __syncthreads();

#pragma unroll 1
  for (int t = 0; t < TT; ++t) {
    // ---- A: load observation column + build history concat for u_t ----
    if (tid < DD) {
      s_x[tid]  = p.data[0*DD*TT + tid*TT + t];
      s_m[tid]  = p.data[1*DD*TT + tid*TT + t];
      s_dl[tid] = p.data[2*DD*TT + tid*TT + t];
    }
    if (tid >= 256 && tid < 256 + H3) {
      int j = tid - 256;
      if (t == 1) {
        if (j < 128) hcat[j] = (j < 64) ? p.h0[j] : hist[0][j - 64];
      } else if (t == 2) {
        hcat[j] = (j < 64) ? p.h0[j] : ((j < 128) ? hist[0][j - 64] : hist[1][j - 128]);
      } else if (t >= 3) {
        hcat[j] = hist[(t - 3 + (j >> 6)) % 3][j & 63];
      }
    }
    __syncthreads();
    // ---- B: gamma_h decay of h; x_st imputation ----
    if (tid < HIDN) {
      float acc = p.gh_b[tid];
      for (int dd = 0; dd < DD; ++dd) acc += p.gh_w[tid*DD + dd] * s_dl[dd];
      sh[tid] *= __expf(-fmaxf(acc, 0.0f));
    } else if (tid >= 64 && tid < 64 + DD) {
      int dd = tid - 64;
      float acc = p.xst_b[0];
      for (int i = 0; i < HIDN; ++i) acc += sH[dd*HIDN + i] * p.xst_w[i];
      s_xst[dd]  = acc;
      s_xstc[dd] = s_m[dd]*s_x[dd] + (1.0f - s_m[dd])*acc;
    }
    __syncthreads();
    // ---- C: GRU gate pre-activations ----
    if (tid < H3) {
      float acc = p.gru_bih[tid];
      for (int dd = 0; dd < DD; ++dd) acc += p.gru_wih[tid*DD + dd] * s_xstc[dd];
      s_gi[tid] = acc;
    } else if (tid >= 256 && tid < 256 + H3) {
      int j = tid - 256;
      float acc = p.gru_bhh[j];
      for (int i = 0; i < HIDN; ++i) acc += p.gru_whh[j*HIDN + i] * sh[i];
      s_gh[j] = acc;
    }
    __syncthreads();
    // ---- D: GRU combine (new h); u_t from history (independent of new h) ----
    if (tid < HIDN) {
      float r = sigm(s_gi[tid]        + s_gh[tid]);
      float z = sigm(s_gi[64 + tid]   + s_gh[64 + tid]);
      float n = tanhf(s_gi[128 + tid] + r * s_gh[128 + tid]);
      sh[tid] = (1.0f - z)*n + z*sh[tid];
    } else if (tid >= 64 && tid < 128) {
      int q = tid - 64;
      float acc;
      if (t == 0) {
        acc = p.wu0_b[q];
        for (int i = 0; i < HIDN; ++i) acc += p.wu0_w[q*HIDN + i] * p.h0[i];
      } else if (t == 1) {
        acc = p.wu1_b[q];
        for (int i = 0; i < 128; ++i) acc += p.wu1_w[q*128 + i] * hcat[i];
      } else {
        acc = p.wu_b[q];
        for (int i = 0; i < H3; ++i) acc += p.wu_w[q*H3 + i] * hcat[i];
      }
      s_u[q] = acc;
    }
    __syncthreads();
    // ---- E: x_gt_hat; gamma_st ----
    if (tid < DD) {
      float acc = p.wh_b[tid];
      for (int i = 0; i < HIDN; ++i) acc += p.wh_w[tid*128 + i]      * sh[i];
      for (int i = 0; i < QQ;  ++i)  acc += p.wh_w[tid*128 + 64 + i] * s_u[i];
      s_xgt[tid] = acc;
    } else if (tid >= 64 && tid < 128) {
      int i = tid - 64;
      float acc = p.gst_b[i];
      for (int dd = 0; dd < DD; ++dd) acc += p.gst_w[i*DD + dd] * s_dl[dd];
      s_gamma[i] = __expf(-fmaxf(acc, 0.0f));
    }
    __syncthreads();
    // ---- F: decay H ----
    for (int e = tid; e < DD*HIDN; e += blockDim.x) sH[e] *= s_gamma[e & 63];
    __syncthreads();
    // ---- G: tensor-GRU batched matvecs via WMMA (f16 in, f32 accum) ----
    {
      int wave = tid >> 5;
      unsigned lane = tid & 31u;
      const v16h* fb = (const v16h*)p.wpk;   // pre-packed fragments (may be null)
      for (int task = wave; task < DD*16; task += 32) {   // wave-uniform loop
        int mt   = task & 3;
        int rest = task >> 2;
        int type = rest & 3;
        int d    = rest >> 2;
        // tensor ids in packed order {tWz=0,tUzh=1,tWr=2,tUrh=3,tWh=4,tUhh=5}
        int t1, t2 = -1;
        const float *v1, *v2 = nullptr;
        float* dst;
        switch (type) {
          case 0:  t1 = 0; t2 = 1; v1 = &sH[d*HIDN]; v2 = sh; dst = sZ;  break;
          case 1:  t1 = 2; t2 = 3; v1 = &sH[d*HIDN]; v2 = sh; dst = sR;  break;
          case 2:  t1 = 4;         v1 = &sH[d*HIDN];          dst = sWh; break;
          default: t1 = 5;         v1 = sh;                   dst = sUh; break;
        }
        int row0 = mt * 16;
        v8f acc = {};
        if (fb) {
          // prefetch next round's first fragment toward WGP while we compute
          int ntask = task + 32;
          if (ntask < DD*16) {
            int nmt = ntask & 3; int nrest = ntask >> 2;
            int ntp = nrest & 3; int ndd = nrest >> 2;
            int nt1 = (ntp == 0) ? 0 : (ntp == 1) ? 2 : (ntp == 2) ? 4 : 5;
            int nf  = ((nt1*DD + ndd)*4 + nmt)*2;
            __builtin_prefetch((const void*)(fb + (size_t)nf*32 + lane), 0, 0);
          }
#pragma unroll
          for (int kt = 0; kt < 2; ++kt) {
            int f = ((t1*DD + d)*4 + mt)*2 + kt;
            v16h a = fb[(size_t)f*32 + lane];
            v16h b = load_B_vec(v1, kt*32);
            acc = __builtin_amdgcn_wmma_f32_16x16x32_f16(false, a, false, b,
                                                         (short)0, acc, false, false);
          }
          if (t2 >= 0) {
#pragma unroll
            for (int kt = 0; kt < 2; ++kt) {
              int f = ((t2*DD + d)*4 + mt)*2 + kt;
              v16h a = fb[(size_t)f*32 + lane];
              v16h b = load_B_vec(v2, kt*32);
              acc = __builtin_amdgcn_wmma_f32_16x16x32_f16(false, a, false, b,
                                                           (short)0, acc, false, false);
            }
          }
        } else {
          // fallback: gather+convert from f32 tensors on the fly
          const float* T1s[6] = { p.tWz, p.tUzh, p.tWr, p.tUrh, p.tWh, p.tUhh };
          const float* W1 = T1s[t1] + d*HIDN*HIDN;
#pragma unroll
          for (int kt = 0; kt < 2; ++kt) {
            v16h a = load_A_frag_f32(W1, row0, kt*32);
            v16h b = load_B_vec(v1, kt*32);
            acc = __builtin_amdgcn_wmma_f32_16x16x32_f16(false, a, false, b,
                                                         (short)0, acc, false, false);
          }
          if (t2 >= 0) {
            const float* W2 = T1s[t2] + d*HIDN*HIDN;
#pragma unroll
            for (int kt = 0; kt < 2; ++kt) {
              v16h a = load_A_frag_f32(W2, row0, kt*32);
              v16h b = load_B_vec(v2, kt*32);
              acc = __builtin_amdgcn_wmma_f32_16x16x32_f16(false, a, false, b,
                                                           (short)0, acc, false, false);
            }
          }
        }
        // Result column N=0 lives in lanes 0 (M=row0..row0+7) and 16 (M=row0+8..+15)
        if ((lane & 15u) == 0u) {
          int mb = d*HIDN + row0 + ((lane < 16u) ? 0 : 8);
#pragma unroll
          for (int r2 = 0; r2 < 8; ++r2) dst[mb + r2] = acc[r2];
        }
      }
    }
    __syncthreads();
    // ---- H: tensor-GRU elementwise combine; write H_s[t] ----
    for (int e = tid; e < DD*HIDN; e += blockDim.x) {
      int i = e & 63;
      float xr = s_xstc[e >> 6];
      float z  = sigm(sZ[e] + p.tUzx[e]*xr + p.tbz[i]);
      float r  = sigm(sR[e] + p.tUrx[e]*xr + p.tbr[i]);
      float hh = tanhf(r*sWh[e] + p.tUhx[e]*xr + sUh[e] + p.tbh[i]);
      float Hn = z*sH[e] + (1.0f - z)*hh;
      sH[e] = Hn;
      p.out[t*DD*HIDN + e] = Hn;
    }
    __syncthreads();
    // ---- I: attention logits; push h into history ring ----
    if (tid < DD) {
      float aa = 0.0f, bb = 0.0f;
      for (int i = 0; i < HIDN; ++i) {
        aa += sh[i] * p.wrg_w[tid*HIDN + i];
        bb += p.wrs[i] * sH[tid*HIDN + i];
      }
      s_a[tid] = aa; s_b[tid] = bb;
    } else if (tid >= 64 && tid < 128) {
      hist[t % 3][tid - 64] = sh[tid - 64];
    }
    __syncthreads();
    // ---- J: dual softmax over D=33 + x_pred column (tiny; serial) ----
    if (tid == 0) {
      float c1[DD], c2[DD];
      float m1 = -1e30f, m2 = -1e30f;
      for (int dd = 0; dd < DD; ++dd) {
        float aa = s_a[dd], bb = s_b[dd], s = aa + bb;
        c1[dd] = aa / s; c2[dd] = bb / s;
        m1 = fmaxf(m1, c1[dd]); m2 = fmaxf(m2, c2[dd]);
      }
      float s1 = 0.0f, s2 = 0.0f;
      for (int dd = 0; dd < DD; ++dd) {
        c1[dd] = __expf(c1[dd] - m1); s1 += c1[dd];
        c2[dd] = __expf(c2[dd] - m2); s2 += c2[dd];
      }
      for (int dd = 0; dd < DD; ++dd) {
        p.out[HSZ + dd*TT + t] = (c1[dd]/s1)*s_xgt[dd] + (c2[dd]/s2)*s_xst[dd];
      }
    }
    __syncthreads();
  }
}

extern "C" void kernel_launch(void* const* d_in, const int* in_sizes, int n_in,
                              void* d_out, int out_size, void* d_ws, size_t ws_size,
                              hipStream_t stream) {
  (void)in_sizes; (void)n_in; (void)out_size;
  Params p;
  p.data    = (const float*)d_in[0];
  p.h0      = (const float*)d_in[1];
  p.H0      = (const float*)d_in[2];
  p.gh_w    = (const float*)d_in[3];
  p.gh_b    = (const float*)d_in[4];
  p.gst_w   = (const float*)d_in[5];
  p.gst_b   = (const float*)d_in[6];
  p.gru_wih = (const float*)d_in[7];
  p.gru_whh = (const float*)d_in[8];
  p.gru_bih = (const float*)d_in[9];
  p.gru_bhh = (const float*)d_in[10];
  p.wu0_w   = (const float*)d_in[11];
  p.wu0_b   = (const float*)d_in[12];
  p.wu1_w   = (const float*)d_in[13];
  p.wu1_b   = (const float*)d_in[14];
  p.wu_w    = (const float*)d_in[15];
  p.wu_b    = (const float*)d_in[16];
  p.wh_w    = (const float*)d_in[17];
  p.wh_b    = (const float*)d_in[18];
  p.xst_w   = (const float*)d_in[19];
  p.xst_b   = (const float*)d_in[20];
  p.wrg_w   = (const float*)d_in[21];
  p.wrs     = (const float*)d_in[22];
  p.tWz     = (const float*)d_in[23];
  p.tUzx    = (const float*)d_in[24];
  p.tUzh    = (const float*)d_in[25];
  p.tWr     = (const float*)d_in[26];
  p.tUrx    = (const float*)d_in[27];
  p.tUrh    = (const float*)d_in[28];
  p.tWh     = (const float*)d_in[29];
  p.tUhx    = (const float*)d_in[30];
  p.tUhh    = (const float*)d_in[31];
  p.tbz     = (const float*)d_in[32];
  p.tbr     = (const float*)d_in[33];
  p.tbh     = (const float*)d_in[34];
  p.out     = (float*)d_out;

  const size_t need = (size_t)NFRAG * 512 * sizeof(_Float16);   // 1.58 MB
  if (d_ws != nullptr && ws_size >= need) {
    p.wpk = d_ws;
    int threads = NFRAG * 32;
    gmgru_pack_wf16<<<dim3((threads + 255) / 256), dim3(256), 0, stream>>>(
        p.tWz, p.tUzh, p.tWr, p.tUrh, p.tWh, p.tUhh, (_Float16*)d_ws);
  } else {
    p.wpk = nullptr;
  }

  gmgru_persistent<<<dim3(1), dim3(1024), 0, stream>>>(p);
}
